// SioConvNet_33294586478863
// MI455X (gfx1250) — matrix operand order repaired and checked by reading, not compiled, not run on hardware
//
#include <hip/hip_runtime.h>

// ---------------------------------------------------------------------------
// SioConvNet forward for MI455X (gfx1250, wave32).
//  * All GEMMs via v_wmma_f32_16x16x32_bf16, fp32 accumulate.
//  * Operands pre-converted to bf16 in workspace when ws_size allows.
//  * 4 waves/block, 32x32 C-tile per wave, BK=64, double-buffered LDS.
//  * Tile staging via Tensor Data Mover (tensor_load_to_lds + s_wait_tensorcnt)
//    when available; TDM pad feature produces the bank-conflict-free LDS stride.
//  * Split-K (grid.z) + deterministic reduction for the 1024x512xK=32000 GEMM
//    (otherwise only 128 workgroups -> underfilled machine).
//  * The O(L^2) recurrence is algebraically a backward complex scan.
// ---------------------------------------------------------------------------

#define DEPTH 3
#define DIM   512
#define DFF   2048
#define DT    64
#define VOCAB 32000
#define BB    2
#define LL    512
#define MROWS (BB * LL)   // 1024 token rows
#define KSPLIT 10         // split-K chunks for the token-in GEMM (32000/10=3200)

typedef __attribute__((ext_vector_type(16))) __bf16        v16bf;
typedef __attribute__((ext_vector_type(8)))  float         v8f;
typedef __attribute__((ext_vector_type(4)))  float         v4f;
typedef __attribute__((ext_vector_type(4)))  unsigned int  v4u;
typedef __attribute__((ext_vector_type(4)))  int           v4i;
typedef __attribute__((ext_vector_type(8)))  int           v8i;

union FragU { v16bf v; v4u u[2]; };

#if __has_builtin(__builtin_amdgcn_tensor_load_to_lds)
#define HAVE_TDM 1
#else
#define HAVE_TDM 0
#endif

#define ATH_STR2(x) #x
#define ATH_STR(x) ATH_STR2(x)
#pragma message("ATHENA-PROBE clang=" ATH_STR(__clang_major__) " HAVE_TDM=" ATH_STR(HAVE_TDM))

__device__ __forceinline__ unsigned short f2bf(float f) {
  __bf16 h = (__bf16)f;                       // hardware f32->bf16 convert
  return __builtin_bit_cast(unsigned short, h);
}
__device__ __forceinline__ unsigned pack2(float lo, float hi) {
  return (unsigned)f2bf(lo) | ((unsigned)f2bf(hi) << 16);
}

// ---------------------------------------------------------------------------
// GEMM: C[M,N] = act(A[M,Ks] @ B[N,Ks]^T + bias[N]) (+ resid). A,B K-major.
// blockIdx.z selects a K-chunk of length Klen (kOff = z*Klen) and writes to
// C + z*M*N; with gridDim.z==1 this is a plain GEMM.
// ---------------------------------------------------------------------------
#define BMg 64
#define BNg 64
#define BKg 64
#define RSg 72    // padded LDS row stride in ushorts (+16B) -> conflict-free

// One 64x64 bf16 tile DMA: global (K-major) -> LDS, pad_enable inserts 16B
// after every 128B row to produce the RSg stride.
__device__ __forceinline__ void tdm_issue_tile(unsigned lds_off,
                                               const unsigned short* tile,
                                               int Kstride, int rowsTotal)
{
#if HAVE_TDM
  unsigned long long ga = (unsigned long long)(const void*)tile;
  v4u g0;
  g0.x = 1u;                                   // count=1, user mode
  g0.y = lds_off;                              // lds_addr
  g0.z = (unsigned)ga;                         // global_addr[31:0]
  g0.w = (unsigned)(ga >> 32) | (2u << 30);    // global_addr[56:32] | type=2
  v8i g1;
  // data_size=2B | pad_enable | pad_interval=128B (code 4) | pad_amount=16B (code 3)
  g1[0] = (int)((1u << 16) | (1u << 20) | (4u << 22) | (3u << 25));
  g1[1] = (int)(((unsigned)Kstride & 0xFFFFu) << 16);                 // tensor_dim0 lo
  g1[2] = (int)(((unsigned)Kstride >> 16) |
                (((unsigned)rowsTotal & 0xFFFFu) << 16));             // dim0 hi | dim1 lo
  g1[3] = (int)(((unsigned)rowsTotal >> 16) | ((unsigned)BKg << 16)); // dim1 hi | tile_dim0
  g1[4] = (int)BMg;                            // tile_dim1=64 rows, tile_dim2=0
  g1[5] = Kstride;                             // tensor_dim0_stride (elements)
  g1[6] = 0;
  g1[7] = 0;
  v4i z4 = {0, 0, 0, 0};
#if __clang_major__ >= 23
  v8i z8 = {0, 0, 0, 0, 0, 0, 0, 0};
  __builtin_amdgcn_tensor_load_to_lds(g0, g1, z4, z4, z8, 0);
#else
  __builtin_amdgcn_tensor_load_to_lds(g0, g1, z4, z4, 0);
#endif
#endif
}

__device__ __forceinline__ void wait_tensor0() {
#if HAVE_TDM
  __builtin_amdgcn_s_wait_tensorcnt(0);
#endif
}

// VALU staging path (bf16 copy, or fp32 load + hw convert).
template<bool BF>
__device__ __forceinline__ void stage_tile(unsigned short (*dst)[RSg], const void* base,
                                           int Kstride, int rowBase, int k0, int lr, int lc)
{
  if (BF) {
    const unsigned short* s = (const unsigned short*)base + (size_t)(rowBase + lr) * Kstride + k0 + lc;
    __builtin_prefetch(s + BKg, 0, 3);
    #pragma unroll
    for (int q = 0; q < 4; ++q)
      *(v4u*)&dst[lr][lc + q * 8] = *(const v4u*)(s + q * 8);
  } else {
    const float* s = (const float*)base + (size_t)(rowBase + lr) * Kstride + k0 + lc;
    __builtin_prefetch(s + BKg, 0, 3);
    #pragma unroll
    for (int q = 0; q < 4; ++q) {
      v4f f0 = *(const v4f*)(s + q * 8);
      v4f f1 = *(const v4f*)(s + q * 8 + 4);
      v4u p = { pack2(f0.x, f0.y), pack2(f0.z, f0.w),
                pack2(f1.x, f1.y), pack2(f1.z, f1.w) };
      *(v4u*)&dst[lr][lc + q * 8] = p;
    }
  }
}

template<bool ABF, bool BBF, bool TDM>
__global__ __launch_bounds__(128)
void gemm_wmma(const void* __restrict__ A, const void* __restrict__ B,
               const float* __restrict__ bias, const float* __restrict__ resid,
               float* __restrict__ C, int M, int N, int Kstride, int Klen, int act)
{
  __shared__ unsigned short As[2][BMg][RSg];   // 18 KB
  __shared__ unsigned short Bs[2][BNg][RSg];   // 18 KB

  const int tid  = threadIdx.x;
  const int lane = tid & 31;
  const int wave = tid >> 5;
  const int bm   = blockIdx.x * BMg;
  const int bn   = blockIdx.y * BNg;
  const int kOff = blockIdx.z * Klen;
  C += (size_t)blockIdx.z * M * N;             // split-K partial slice
  const int wm   = (wave & 1) * 32;
  const int wn   = (wave >> 1) * 32;
  const int lr   = tid >> 1;                   // staging row (0..63)
  const int lc   = (tid & 1) * 32;             // staging col base (0/32)

  const v8f zero = {0.f, 0.f, 0.f, 0.f, 0.f, 0.f, 0.f, 0.f};
  v8f acc[2][2];
  acc[0][0] = zero; acc[0][1] = zero; acc[1][0] = zero; acc[1][1] = zero;

  const bool tdmA = TDM && ABF;
  const bool tdmB = TDM && BBF;

  auto issue = [&](int buf, int k0) {
    if ((tdmA || tdmB) && wave == 0) {
      if (tdmA) tdm_issue_tile((unsigned)(size_t)&As[buf][0][0],
                               (const unsigned short*)A + (size_t)bm * Kstride + kOff + k0,
                               Kstride, M);
      if (tdmB) tdm_issue_tile((unsigned)(size_t)&Bs[buf][0][0],
                               (const unsigned short*)B + (size_t)bn * Kstride + kOff + k0,
                               Kstride, N);
    }
    if (!tdmA) stage_tile<ABF>(As[buf], A, Kstride, bm, kOff + k0, lr, lc);
    if (!tdmB) stage_tile<BBF>(Bs[buf], B, Kstride, bn, kOff + k0, lr, lc);
  };

  const int nk = Klen / BKg;
  issue(0, 0);
  if ((tdmA || tdmB) && wave == 0) wait_tensor0();
  __syncthreads();

  for (int t = 0; t < nk; ++t) {
    const int p = t & 1;
    if (t + 1 < nk) issue(1 - p, (t + 1) * BKg);   // fill other buffer

    #pragma unroll
    for (int kk = 0; kk < BKg; kk += 32) {
      FragU fa[2], fb[2];
      // A frag (16x32): lanes 0-15 M=lane, K kk+[0..7]&[16..23]; lanes 16-31 K+8.
      #pragma unroll
      for (int mh = 0; mh < 2; ++mh) {
        const unsigned short* ap = &As[p][wm + mh * 16 + (lane & 15)][kk + (lane >> 4) * 8];
        fa[mh].u[0] = *(const v4u*)(ap);
        fa[mh].u[1] = *(const v4u*)(ap + 16);
      }
      // B frag (32x16): lane n -> column n&15, K kk+[0..7]&[16..23]; lanes 16-31 K+8.
      #pragma unroll
      for (int nh = 0; nh < 2; ++nh) {
        const unsigned short* bp = &Bs[p][wn + nh * 16 + (lane & 15)][kk + (lane >> 4) * 8];
        fb[nh].u[0] = *(const v4u*)(bp);
        fb[nh].u[1] = *(const v4u*)(bp + 16);
      }
      #pragma unroll
      for (int mh = 0; mh < 2; ++mh)
        #pragma unroll
        for (int nh = 0; nh < 2; ++nh)
          acc[mh][nh] = __builtin_amdgcn_wmma_f32_16x16x32_bf16(
              false, fa[mh].v, false, fb[nh].v, (short)0, acc[mh][nh], false, false);
    }

    if ((tdmA || tdmB) && wave == 0 && t + 1 < nk) wait_tensor0();
    __syncthreads();
  }

  // Epilogue. C/D layout: VGPR j -> M = j + 8*(lane>>4), N = lane&15.
  #pragma unroll
  for (int mh = 0; mh < 2; ++mh) {
    const int rbase = bm + wm + mh * 16 + ((lane >> 4) << 3);
    #pragma unroll
    for (int nh = 0; nh < 2; ++nh) {
      const int col = bn + wn + nh * 16 + (lane & 15);
      const float bia = bias ? bias[col] : 0.f;
      #pragma unroll
      for (int j = 0; j < 8; ++j) {
        const int row = rbase + j;
        float v = acc[mh][nh][j] + bia;
        if (act) v = v / (1.f + __expf(-v));     // SiLU
        const size_t i0 = (size_t)row * N + col;
        if (resid) v += resid[i0];
        C[i0] = v;
      }
    }
  }
}

// ---------------------------------------------------------------------------
// Deterministic split-K reduction: C[i] = bias[i%N] + sum_s part[s][i].
// ---------------------------------------------------------------------------
__global__ __launch_bounds__(256)
void reduce_k(const float* __restrict__ part, const float* __restrict__ bias,
              float* __restrict__ C, int MN, int N, int S)
{
  const int i = blockIdx.x * 256 + threadIdx.x;
  if (i >= MN) return;
  float s = bias ? bias[i % N] : 0.f;
  for (int k = 0; k < S; ++k) s += part[(size_t)k * MN + i];
  C[i] = s;
}

// ---------------------------------------------------------------------------
// fp32 -> bf16 bulk convert (8 elems/thread, hw cvt, 128-bit stores).
// ---------------------------------------------------------------------------
__global__ __launch_bounds__(256)
void cvt_bf16(const float* __restrict__ src, unsigned short* __restrict__ dst, long n)
{
  const long i = ((long)blockIdx.x * 256 + threadIdx.x) * 8;
  if (i >= n) return;
  v4f a = *(const v4f*)(src + i);
  v4f b = *(const v4f*)(src + i + 4);
  v4u p = { pack2(a.x, a.y), pack2(a.z, a.w), pack2(b.x, b.y), pack2(b.z, b.w) };
  *(v4u*)(dst + i) = p;
}

// ---------------------------------------------------------------------------
// LayerNorm over DIM=512: one wave32 per row.
// ---------------------------------------------------------------------------
__global__ __launch_bounds__(256)
void ln_rows(const float* __restrict__ x, const float* __restrict__ g,
             const float* __restrict__ b, float* __restrict__ o)
{
  const int row  = blockIdx.x * 8 + (threadIdx.x >> 5);
  const int lane = threadIdx.x & 31;
  const float* xr = x + (size_t)row * DIM;
  float v[16];
  float s = 0.f;
  #pragma unroll
  for (int j = 0; j < 16; ++j) { v[j] = xr[lane + 32 * j]; s += v[j]; }
  #pragma unroll
  for (int off = 16; off > 0; off >>= 1) s += __shfl_xor(s, off, 32);
  const float mean = s * (1.f / DIM);
  float sv = 0.f;
  #pragma unroll
  for (int j = 0; j < 16; ++j) { const float d = v[j] - mean; sv += d * d; }
  #pragma unroll
  for (int off = 16; off > 0; off >>= 1) sv += __shfl_xor(sv, off, 32);
  const float inv = rsqrtf(sv * (1.f / DIM) + 1e-5f);
  float* orow = o + (size_t)row * DIM;
  #pragma unroll
  for (int j = 0; j < 16; ++j) {
    const int c = lane + 32 * j;
    orow[c] = (v[j] - mean) * inv * g[c] + b[c];
  }
}

// ---------------------------------------------------------------------------
// SioConv recurrence as a backward complex scan per (b,d); fused gate.
// ---------------------------------------------------------------------------
__global__ __launch_bounds__(128)
void sioconv_scan(const float* __restrict__ u, const float* __restrict__ za,
                  const float* __restrict__ h0, const float* __restrict__ y,
                  float* __restrict__ out)
{
  const int t = threadIdx.x;          // 0..127
  const int b = t >> 6;
  const int d = t & (DT - 1);
  const float* ub = u   + (size_t)b * LL * DT;
  const float* zb = za  + (size_t)b * LL * 2 * DT;
  const float* yb = y   + (size_t)b * LL * DT;
  float*       ob = out + (size_t)b * LL * DT;
  const float h0r = h0[2 * d + 0];
  const float h0i = h0[2 * d + 1];

  float gr = 0.f, gi = 0.f, par = 0.f, pai = 0.f;
  for (int i = LL - 1; i >= 0; --i) {
    const float za0 = zb[(size_t)i * 2 * DT + 2 * d + 0];
    const float za1 = zb[(size_t)i * 2 * DT + 2 * d + 1];
    const float m2  = za0 * za0 + za1 * za1;
    const float sc  = rsqrtf(m2) * __expf(-m2);
    const float ar  = za0 * sc, ai = za1 * sc;

    const float xri = (i == 0) ? h0r : ub[(size_t)(i - 1) * DT + d];
    const float xii = (i == 0) ? h0i : 0.f;
    if (i == LL - 1) { gr = xri; gi = xii; }
    else {
      const float ngr = xri + par * gr - pai * gi;
      const float ngi = xii + par * gi + pai * gr;
      gr = ngr; gi = ngi;
    }
    float hr = ar * gr - ai * gi;
    if (i > 0) hr += (i == 1) ? h0r : ub[(size_t)(i - 2) * DT + d];
    if (i == LL - 1) hr += ub[(size_t)i * DT + d];
    ob[(size_t)i * DT + d] = hr * yb[(size_t)i * DT + d];
    par = ar; pai = ai;
  }
}

// ---------------------------------------------------------------------------
// Orchestration
// ---------------------------------------------------------------------------
extern "C" void kernel_launch(void* const* d_in, const int* in_sizes, int n_in,
                              void* d_out, int out_size, void* d_ws, size_t ws_size,
                              hipStream_t stream)
{
  const float* x         = (const float*)d_in[0];
  const float* tok_in_w  = (const float*)d_in[1];
  const float* tok_in_b  = (const float*)d_in[2];
  const float* ln1_g     = (const float*)d_in[3];
  const float* ln1_b     = (const float*)d_in[4];
  const float* fc_w      = (const float*)d_in[5];
  const float* fc_b      = (const float*)d_in[6];
  const float* tin_w     = (const float*)d_in[7];
  const float* tin_b     = (const float*)d_in[8];
  const float* a_w       = (const float*)d_in[9];
  const float* h0p       = (const float*)d_in[10];
  const float* tout_w    = (const float*)d_in[11];
  const float* tout_b    = (const float*)d_in[12];
  const float* ln2_g     = (const float*)d_in[13];
  const float* ln2_b     = (const float*)d_in[14];
  const float* ffn_w1    = (const float*)d_in[15];
  const float* ffn_b1    = (const float*)d_in[16];
  const float* ffn_w2    = (const float*)d_in[17];
  const float* ffn_b2    = (const float*)d_in[18];
  const float* lnf_g     = (const float*)d_in[19];
  const float* lnf_b     = (const float*)d_in[20];
  const float* tok_out_w = (const float*)d_in[21];
  const float* tok_out_b = (const float*)d_in[22];
  float* out = (float*)d_out;

  // ---- workspace layout: fp32 core | split-K partials | bf16 region ------
  float* wsf = (float*)d_ws;
  float* h   = wsf; wsf += MROWS * DIM;
  float* xn  = wsf; wsf += MROWS * DIM;
  float* yb  = wsf; wsf += MROWS * DT;
  float* ub  = wsf; wsf += MROWS * DT;
  float* za  = wsf; wsf += MROWS * 2 * DT;
  float* gb  = wsf; wsf += MROWS * DT;
  float* f1  = wsf; wsf += MROWS * DFF;
  float* part = wsf; wsf += (size_t)KSPLIT * MROWS * DIM;       // 21 MB
  const size_t need_split = (char*)wsf - (char*)d_ws;

  unsigned short* wsh = (unsigned short*)wsf;
  unsigned short* xnb   = wsh; wsh += MROWS * DIM;
  unsigned short* ubb   = wsh; wsh += MROWS * DT;
  unsigned short* gbb   = wsh; wsh += MROWS * DT;
  unsigned short* f1b   = wsh; wsh += MROWS * DFF;
  unsigned short* tinwB = wsh; wsh += (size_t)DIM * VOCAB;      // tok_in_w
  unsigned short* toutwB= wsh; wsh += (size_t)VOCAB * DIM;      // tok_out_w
  unsigned short* fcB   = wsh; wsh += (size_t)DEPTH * DT * DIM;
  unsigned short* tinB  = wsh; wsh += (size_t)DEPTH * DT * DIM;
  unsigned short* awB   = wsh; wsh += (size_t)DEPTH * 2 * DT * DT;
  unsigned short* towB  = wsh; wsh += (size_t)DEPTH * DIM * DT;
  unsigned short* w1B   = wsh; wsh += (size_t)DEPTH * DFF * DIM;
  unsigned short* w2B   = wsh; wsh += (size_t)DEPTH * DIM * DFF;
  const size_t need_mid = (char*)wsh - (char*)d_ws;             // + bf16 weights/acts
  unsigned short* xB    = wsh; wsh += (size_t)MROWS * VOCAB;
  const size_t need_full = (char*)wsh - (char*)d_ws;            // + x in bf16

  const bool doSplit = ws_size >= need_split;
  const bool tier1   = ws_size >= need_mid;    // bf16 weights + activations
  const bool tier2   = ws_size >= need_full;   // + bf16 copy of x

  const dim3 b128(128), b256(256);
  constexpr bool TDM = (HAVE_TDM != 0);

  auto gemmF = [&](float* C, const float* Af, const float* Bf, const float* bias,
                   const float* resid, int M, int N, int K, int act) {
    gemm_wmma<false, false, false><<<dim3(M / 64, N / 64), b128, 0, stream>>>(
        Af, Bf, bias, resid, C, M, N, K, K, act);
  };
  auto gemmM = [&](float* C, const float* Af, const unsigned short* Bh, const float* bias,
                   const float* resid, int M, int N, int K, int act) {
    gemm_wmma<false, true, TDM><<<dim3(M / 64, N / 64), b128, 0, stream>>>(
        Af, Bh, bias, resid, C, M, N, K, K, act);
  };
  auto gemmH = [&](float* C, const unsigned short* Ah, const unsigned short* Bh,
                   const float* bias, const float* resid, int M, int N, int K, int act) {
    gemm_wmma<true, true, TDM><<<dim3(M / 64, N / 64), b128, 0, stream>>>(
        Ah, Bh, bias, resid, C, M, N, K, K, act);
  };
  auto cvt = [&](const float* s, unsigned short* d, long n) {
    cvt_bf16<<<(int)((n / 8 + 255) / 256), b256, 0, stream>>>(s, d, n);
  };

  if (tier1) {                       // per-launch weight conversion (deterministic)
    cvt(tok_in_w,  tinwB,  (long)DIM * VOCAB);
    cvt(tok_out_w, toutwB, (long)VOCAB * DIM);
    cvt(fc_w,   fcB,  (long)DEPTH * DT * DIM);
    cvt(tin_w,  tinB, (long)DEPTH * DT * DIM);
    cvt(a_w,    awB,  (long)DEPTH * 2 * DT * DT);
    cvt(tout_w, towB, (long)DEPTH * DIM * DT);
    cvt(ffn_w1, w1B,  (long)DEPTH * DFF * DIM);
    cvt(ffn_w2, w2B,  (long)DEPTH * DIM * DFF);
    if (tier2) cvt(x, xB, (long)MROWS * VOCAB);
  }

  // Token-in projection (1024 x 32000 x 512): split-K to fill the machine.
  if (doSplit) {
    const dim3 gs(MROWS / 64, DIM / 64, KSPLIT);
    const int klen = VOCAB / KSPLIT;          // 3200 = 50 BK tiles
    if (tier2)
      gemm_wmma<true, true, TDM><<<gs, b128, 0, stream>>>(
          xB, tinwB, nullptr, nullptr, part, MROWS, DIM, VOCAB, klen, 0);
    else if (tier1)
      gemm_wmma<false, true, TDM><<<gs, b128, 0, stream>>>(
          x, tinwB, nullptr, nullptr, part, MROWS, DIM, VOCAB, klen, 0);
    else
      gemm_wmma<false, false, false><<<gs, b128, 0, stream>>>(
          x, tok_in_w, nullptr, nullptr, part, MROWS, DIM, VOCAB, klen, 0);
    reduce_k<<<(MROWS * DIM + 255) / 256, b256, 0, stream>>>(
        part, tok_in_b, h, MROWS * DIM, DIM, KSPLIT);
  } else if (tier2) {
    gemmH(h, xB, tinwB, tok_in_b, nullptr, MROWS, DIM, VOCAB, 0);
  } else if (tier1) {
    gemmM(h, x, tinwB, tok_in_b, nullptr, MROWS, DIM, VOCAB, 0);
  } else {
    gemmF(h, x, tok_in_w, tok_in_b, nullptr, MROWS, DIM, VOCAB, 0);
  }

  for (int i = 0; i < DEPTH; ++i) {
    ln_rows<<<MROWS / 8, b256, 0, stream>>>(h, ln1_g + i * DIM, ln1_b + i * DIM, xn);
    if (tier1) {
      cvt(xn, xnb, (long)MROWS * DIM);
      gemmH(yb, xnb, fcB  + (size_t)i * DT * DIM, fc_b  + i * DT, nullptr, MROWS, DT, DIM, 1);
      gemmH(ub, xnb, tinB + (size_t)i * DT * DIM, tin_b + i * DT, nullptr, MROWS, DT, DIM, 0);
      cvt(ub, ubb, (long)MROWS * DT);
      gemmH(za, ubb, awB + (size_t)i * 2 * DT * DT, nullptr, nullptr, MROWS, 2 * DT, DT, 0);
      sioconv_scan<<<1, b128, 0, stream>>>(ub, za, h0p + (size_t)i * DT * 2, yb, gb);
      cvt(gb, gbb, (long)MROWS * DT);
      gemmH(h, gbb, towB + (size_t)i * DIM * DT, tout_b + i * DIM, h, MROWS, DIM, DT, 0);
      ln_rows<<<MROWS / 8, b256, 0, stream>>>(h, ln2_g + i * DIM, ln2_b + i * DIM, xn);
      cvt(xn, xnb, (long)MROWS * DIM);
      gemmH(f1, xnb, w1B + (size_t)i * DFF * DIM, ffn_b1 + i * DFF, nullptr, MROWS, DFF, DIM, 1);
      cvt(f1, f1b, (long)MROWS * DFF);
      gemmH(h, f1b, w2B + (size_t)i * DIM * DFF, ffn_b2 + i * DIM, h, MROWS, DIM, DFF, 0);
    } else {
      gemmF(yb, xn, fc_w  + (size_t)i * DT * DIM, fc_b  + i * DT, nullptr, MROWS, DT, DIM, 1);
      gemmF(ub, xn, tin_w + (size_t)i * DT * DIM, tin_b + i * DT, nullptr, MROWS, DT, DIM, 0);
      gemmF(za, ub, a_w   + (size_t)i * 2 * DT * DT, nullptr,     nullptr, MROWS, 2 * DT, DT, 0);
      sioconv_scan<<<1, b128, 0, stream>>>(ub, za, h0p + (size_t)i * DT * 2, yb, gb);
      gemmF(h, gb, tout_w + (size_t)i * DIM * DT, tout_b + i * DIM, h, MROWS, DIM, DT, 0);
      ln_rows<<<MROWS / 8, b256, 0, stream>>>(h, ln2_g + i * DIM, ln2_b + i * DIM, xn);
      gemmF(f1, xn, ffn_w1 + (size_t)i * DFF * DIM, ffn_b1 + i * DFF, nullptr, MROWS, DFF, DIM, 1);
      gemmF(h, f1, ffn_w2 + (size_t)i * DIM * DFF, ffn_b2 + i * DIM,  h, MROWS, DIM, DFF, 0);
    }
  }

  // Final LN + token-out projection (1024 x 512 x 32000)
  ln_rows<<<MROWS / 8, b256, 0, stream>>>(h, lnf_g, lnf_b, xn);
  if (tier1) {
    cvt(xn, xnb, (long)MROWS * DIM);
    gemmH(out, xnb, toutwB, tok_out_b, nullptr, MROWS, VOCAB, DIM, 0);
  } else {
    gemmF(out, xn, tok_out_w, tok_out_b, nullptr, MROWS, VOCAB, DIM, 0);
  }
}
